// AdaptiveWaveletLayer_71055938945722
// MI455X (gfx1250) — compile-verified, matrix operand order checked by reading, not run on hardware
//
#include <hip/hip_runtime.h>
#include <hip/hip_fp16.h>

constexpr int kB = 8, kN = 512, kT = 12, kC = 64, kHOP = 3;
constexpr float kLeak = 0.2f;
constexpr int STRD = 520; // padded halves per transposed row: 16B aligned, bank-skewed

typedef __attribute__((ext_vector_type(16))) _Float16 v16h;
typedef __attribute__((ext_vector_type(8)))  _Float16 v8h;
typedef __attribute__((ext_vector_type(8)))  float    v8f;

struct __align__(16) SharedBlk {
  _Float16 xT[kC * STRD];    // x transposed [c][n], fp16 (matmul src step 0 + feat chain)
  _Float16 updA[kC * STRD];  // update ping
  _Float16 updB[kC * STRD];  // update pong
  unsigned maskW[kN * 16];   // adj>0 bitmask, 32 j's per word
  float f1[kN], f2[kN], mrow[kN], invs[kN], rsum[kN];
  float sA[2 * kC];          // attention vector a
};

__device__ __forceinline__ float sigm(float x) { return 1.0f / (1.0f + __expf(-x)); }

__global__ __launch_bounds__(1024, 1) void gat_cheb_fused(
    const float* __restrict__ input, const float* __restrict__ adj,
    const float* __restrict__ a, const float* __restrict__ temp,
    const float* __restrict__ cheb, float* __restrict__ out) {
  __shared__ SharedBlk s;
  const int tid = threadIdx.x;
  const int bt  = blockIdx.x;
  const int b   = bt / kT, t = bt % kT;
  const float* xg = input + ((size_t)b * kN * kT + t) * kC; // x[n][c] = xg[n*kT*kC + c]

  // ---- phase 1: load a, transpose x -> fp16 LDS, build adj>0 bitmask ----
  if (tid < 2 * kC) s.sA[tid] = a[tid];
  for (int idx = tid; idx < kN * kC; idx += 1024) {
    int n = idx >> 6, c = idx & 63;
    s.xT[c * STRD + n] = (_Float16)xg[(size_t)n * kT * kC + c];
  }
  for (int w = tid; w < kN * 16; w += 1024) {
    int i = w >> 4, jb = w & 15;
    const float4* a4 = (const float4*)(adj + (size_t)i * kN + jb * 32);
    unsigned bits = 0;
#pragma unroll
    for (int u = 0; u < 8; ++u) {
      float4 q = a4[u];
      bits |= (q.x > 0.f ? 1u : 0u) << (u * 4 + 0);
      bits |= (q.y > 0.f ? 1u : 0u) << (u * 4 + 1);
      bits |= (q.z > 0.f ? 1u : 0u) << (u * 4 + 2);
      bits |= (q.w > 0.f ? 1u : 0u) << (u * 4 + 3);
    }
    s.maskW[w] = bits;
  }
  __syncthreads();

  // ---- phase 2: f1 = x@a1, f2 = x@a2 (f32, from global/L2) ----
  if (tid < kN) {
    const float4* row = (const float4*)(xg + (size_t)tid * kT * kC);
    float d1 = 0.f, d2 = 0.f;
#pragma unroll
    for (int c4 = 0; c4 < kC / 4; ++c4) {
      float4 v = row[c4];
      d1 += v.x * s.sA[c4 * 4 + 0] + v.y * s.sA[c4 * 4 + 1] +
            v.z * s.sA[c4 * 4 + 2] + v.w * s.sA[c4 * 4 + 3];
      d2 += v.x * s.sA[kC + c4 * 4 + 0] + v.y * s.sA[kC + c4 * 4 + 1] +
            v.z * s.sA[kC + c4 * 4 + 2] + v.w * s.sA[kC + c4 * 4 + 3];
    }
    s.f1[tid] = d1;
    s.f2[tid] = d2;
  }
  __syncthreads();

  // ---- phase 3: per-row softmax stats: max, 1/sum, rowsum = 0.5*sum(adj*U) ----
  if (tid < kN) {
    const int i = tid;
    const float fi = s.f1[i];
    float mx = -3.402823466e38f;
    for (int jw = 0; jw < 16; ++jw) {
      unsigned w = s.maskW[i * 16 + jw];
      if (!w) continue;
      for (int j2 = 0; j2 < 32; ++j2) {
        if ((w >> j2) & 1u) {
          float e = fi + s.f2[jw * 32 + j2];
          e = e > 0.f ? e : kLeak * e;
          mx = fmaxf(mx, e);
        }
      }
    }
    float ssum = 0.f, rr = 0.f;
    for (int jw = 0; jw < 16; ++jw) {
      unsigned w = s.maskW[i * 16 + jw];
      if (!w) continue;
      for (int j2 = 0; j2 < 32; ++j2) {
        float bitf = (float)((w >> j2) & 1u);
        float e = fi + s.f2[jw * 32 + j2];
        e = e > 0.f ? e : kLeak * e;
        float u = __expf(e - mx) * bitf;
        ssum += u;
        rr += adj[(size_t)i * kN + jw * 32 + j2] * u;
      }
    }
    float inv = 1.0f / ssum;
    s.mrow[i] = mx;
    s.invs[i] = inv;
    s.rsum[i] = 0.5f * rr * inv;
  }
  __syncthreads();

  // ---- hop loop: each wave owns one 16-row i-tile ----
  const float coe0 = sigm(temp[0]), coe1 = sigm(temp[1]), coe2 = sigm(temp[2]);
  const float cc0 = sigm(cheb[0]), cc1 = sigm(cheb[1]);

  const int lane = tid & 31, wave = tid >> 5;
  const int hi = lane >> 4, l15 = lane & 15;
  const int i0 = wave * 16;
  const int rowA = i0 + l15;      // A-operand row owned by this lane
  const int rD = i0 + hi * 8;     // first D/C row owned by this lane
  const int q = hi * 8;           // K base for this lane half (WMMA fp16 layout)
  const float fi = s.f1[rowA];
  const float mi = s.mrow[rowA];

  float fp[4][8]; // feat_prime accumulator: the ONLY persistent per-lane array
#pragma unroll
  for (int step = 0; step < kHOP; ++step) {
    const _Float16* src = (step == 0) ? s.xT : ((step == 1) ? s.updA : s.updB);
    _Float16* dst = (step == 0) ? s.updA : s.updB;
    const float chebf = (step == 0) ? 1.f : ((step == 1) ? cc0 : cc0 * cc1);

    v8f acc[4];
    const v8f vzero = {0.f, 0.f, 0.f, 0.f, 0.f, 0.f, 0.f, 0.f};
#pragma unroll
    for (int ct = 0; ct < 4; ++ct) acc[ct] = vzero;

    for (int jt = 0; jt < 16; ++jt) {
      const int j0 = jt * 32;
      // Batch-load all four B tiles first (8x ds_load_b128, clause-able),
      // then overlap the exp-heavy A build with their latency.
      v16h Bt[4];
#pragma unroll
      for (int ct = 0; ct < 4; ++ct) {
        const _Float16* bp = src + (ct * 16 + l15) * STRD + j0 + q;
        v8h blo = *(const v8h*)bp;          // K = q .. q+7
        v8h bhi = *(const v8h*)(bp + 16);   // K = q+16 .. q+23
        Bt[ct] = __builtin_shufflevector(blo, bhi, 0, 1, 2, 3, 4, 5, 6, 7,
                                         8, 9, 10, 11, 12, 13, 14, 15);
      }
      const unsigned w = s.maskW[rowA * 16 + jt];
      const float4* f2q = (const float4*)(s.f2 + j0 + q);
      float4 fa = f2q[0], fb = f2q[1], fc = f2q[4], fd = f2q[5];
      float ebuf[16] = {fa.x, fa.y, fa.z, fa.w, fb.x, fb.y, fb.z, fb.w,
                        fc.x, fc.y, fc.z, fc.w, fd.x, fd.y, fd.z, fd.w};
      // U-tile A operand in WMMA fp16 register layout:
      // lane row = lane&15; halves h<8 -> K=q+h, h>=8 -> K=q+16+(h-8)
      v16h A;
#pragma unroll
      for (int h = 0; h < 16; ++h) {
        int kk = (h < 8 ? h : h + 8) + q;
        float e = fi + ebuf[h];
        e = e > 0.f ? e : kLeak * e;
        float u = __expf(e - mi);
        u = ((w >> kk) & 1u) ? u : 0.f;
        A[h] = (_Float16)u;
      }
#pragma unroll
      for (int ct = 0; ct < 4; ++ct)
        acc[ct] = __builtin_amdgcn_wmma_f32_16x16x32_f16(
            false, A, false, Bt[ct], (short)0, acc[ct], false, false);
    }

    // ---- epilogue: one c-tile at a time so acc[ct] dies before the next ----
    float inv8[8], rs8[8];
#pragma unroll
    for (int v = 0; v < 8; ++v) inv8[v] = s.invs[rD + v];
#pragma unroll
    for (int v = 0; v < 8; ++v) rs8[v] = s.rsum[rD + v] * chebf;
#pragma unroll
    for (int ct = 0; ct < 4; ++ct) {
      v8h xh = *(const v8h*)(s.xT + (ct * 16 + l15) * STRD + rD);
      v8h pk;
#pragma unroll
      for (int v = 0; v < 8; ++v) {
        float upd = acc[ct][v] * inv8[v]; // update_{k+1} = (U @ update_k)(i,c)
        pk[v] = (_Float16)upd;
        float xv = (float)xh[v];
        float even = coe0 * xv + upd;
        float odd = upd - even * rs8[v];
        float fuse = coe1 * even + (1.f - coe1) * odd;
        float prev = (step == 0) ? xv : fp[ct][v];
        float fnew = coe2 * prev + (1.f - coe2) * fuse;
        if (step == kHOP - 1) {
          // final hop: write result straight out, fp dies here
          out[((size_t)(b * kN + rD + v)) * kT * kC + (size_t)t * kC +
              ct * 16 + l15] = fnew;
        } else {
          fp[ct][v] = fnew;
        }
      }
      if (step < kHOP - 1)
        *(v8h*)(dst + (ct * 16 + l15) * STRD + rD) = pk;
    }
    __syncthreads();
  }
}

extern "C" void kernel_launch(void* const* d_in, const int* in_sizes, int n_in,
                              void* d_out, int out_size, void* d_ws, size_t ws_size,
                              hipStream_t stream) {
  (void)in_sizes; (void)n_in; (void)d_ws; (void)ws_size; (void)out_size;
  const float* input = (const float*)d_in[0];
  /* d_in[1] = h0, unused by the reference */
  const float* adj  = (const float*)d_in[2];
  const float* a    = (const float*)d_in[3];
  const float* temp = (const float*)d_in[4];
  const float* cheb = (const float*)d_in[5];
  float* out = (float*)d_out;
  dim3 grid(kB * kT), block(1024);
  gat_cheb_fused<<<grid, block, 0, stream>>>(input, adj, a, temp, cheb, out);
}
